// HunchLLM_25658134626554
// MI455X (gfx1250) — matrix operand order repaired and checked by reading, not compiled, not run on hardware
//
#include <hip/hip_runtime.h>
#include <hip/hip_bf16.h>

// ---------------------------------------------------------------------------
// Types for CDNA5 WMMA (wave32, 16x16x32 bf16 -> f32)
// ---------------------------------------------------------------------------
typedef __bf16 bf16x16 __attribute__((ext_vector_type(16)));
typedef float  f32x8   __attribute__((ext_vector_type(8)));
typedef unsigned short u16x8  __attribute__((ext_vector_type(8)));
typedef unsigned short u16x16 __attribute__((ext_vector_type(16)));

union BF16FragU { u16x16 u; bf16x16 b; };

// Native bf16 conversions (lowers to v_cvt_*bf16_f32 on gfx1250)
__device__ __forceinline__ unsigned short f2b(float f) {
    union { unsigned short s; __bf16 b; } p;
    p.b = (__bf16)f;
    return p.s;
}
__device__ __forceinline__ unsigned pack2bf(float lo, float hi) {
    union { unsigned u; __bf16 b[2]; } p;
    p.b[0] = (__bf16)lo; p.b[1] = (__bf16)hi;
    return p.u;
}

__device__ __forceinline__ bf16x16 load_frag_contig(const unsigned short* p) {
    BF16FragU f; f.u = *(const u16x16*)p; return f.b;
}
__device__ __forceinline__ bf16x16 load_frag_split(const unsigned short* p1,
                                                   const unsigned short* p2) {
    BF16FragU f;
    u16x8 lo = *(const u16x8*)p1;
    u16x8 hi = *(const u16x8*)p2;
#pragma unroll
    for (int i = 0; i < 8; ++i) { f.u[i] = lo[i]; f.u[i + 8] = hi[i]; }
    return f.b;
}

__device__ __forceinline__ f32x8 wmma_bf16(bf16x16 a, bf16x16 b, f32x8 c) {
    return __builtin_amdgcn_wmma_f32_16x16x32_bf16(
        /*neg_a=*/false, a, /*neg_b=*/false, b,
        /*c_mod=*/(short)0, c, /*reuse_a=*/false, /*reuse_b=*/false);
}

// ---------------------------------------------------------------------------
// Constants
// ---------------------------------------------------------------------------
#define BB   4
#define SRCL 256
#define TGTL 512
#define DM   1024
#define NH   16
#define DK   64
#define NL   8
#define FFD  4096
#define VOC  32000

// ---------------------------------------------------------------------------
// GEMM: C[M,N] = act(A[M,K] @ W[K,N] + bias), all f32 in memory, bf16 WMMA.
// Block tile 128x64, K-step 64, double-buffered LDS (48 KB). 256 threads =
// 8 waves (4x2), each wave a 32x32 patch = 2x2 WMMA accumulators; 8 WMMAs
// per stage per wave with one barrier per 64-deep K step. Global loads for
// stage i+1 are issued before the WMMA burst of stage i (latency hiding).
// Requires M%128==0, N%64==0, K%64==0 (true for all uses here).
// ---------------------------------------------------------------------------
template <int RELU>
__global__ __launch_bounds__(256) void hl_gemm_bias(
    const float* __restrict__ A, const float* __restrict__ W,
    const float* __restrict__ bias, float* __restrict__ C,
    int M, int N, int K)
{
    __shared__ unsigned short As[2][128][64];  // [buf][m][k]
    __shared__ unsigned short Bs[2][64][64];   // [buf][n][k] (transposed)

    const int tid  = threadIdx.x;
    const int lane = tid & 31;
    const int wave = tid >> 5;
    const int wm   = wave >> 1;   // 0..3
    const int wn   = wave & 1;    // 0..1
    const int row0 = blockIdx.y * 128;
    const int col0 = blockIdx.x * 64;

    f32x8 acc[2][2];
    f32x8 zero = {};
    acc[0][0] = zero; acc[0][1] = zero; acc[1][0] = zero; acc[1][1] = zero;

    const int aR = tid >> 4;         // 0..15 (+16*it), A row
    const int aC = (tid & 15) * 4;   // 0..60, A col (k)
    const int bK = (tid >> 4) * 2;   // 0,2,..,30 (+32), W k-pair
    const int bC = (tid & 15) * 4;   // 0..60, W col (n)

    const int klbA = (lane < 16) ? 0 : 8;
    const int klbB = (lane < 16) ? 0 : 16;
    const int nl   = lane & 15;

    float4 aReg[8];
    float4 bReg[2][2];

    auto loadGlobal = [&](int kb) {
#pragma unroll
        for (int it = 0; it < 8; ++it) {
            int r = aR + it * 16;
            aReg[it] = *(const float4*)(A + (size_t)(row0 + r) * K + kb + aC);
        }
#pragma unroll
        for (int it = 0; it < 2; ++it) {
            int kk = bK + it * 32;
            bReg[it][0] = *(const float4*)(W + (size_t)(kb + kk) * N + col0 + bC);
            bReg[it][1] = *(const float4*)(W + (size_t)(kb + kk + 1) * N + col0 + bC);
        }
    };
    auto storeLds = [&](int p) {
#pragma unroll
        for (int it = 0; it < 8; ++it) {
            int r = aR + it * 16;
            uint2 pk;
            pk.x = pack2bf(aReg[it].x, aReg[it].y);
            pk.y = pack2bf(aReg[it].z, aReg[it].w);
            *(uint2*)&As[p][r][aC] = pk;
        }
#pragma unroll
        for (int it = 0; it < 2; ++it) {
            int kk = bK + it * 32;
            *(unsigned*)&Bs[p][bC + 0][kk] = pack2bf(bReg[it][0].x, bReg[it][1].x);
            *(unsigned*)&Bs[p][bC + 1][kk] = pack2bf(bReg[it][0].y, bReg[it][1].y);
            *(unsigned*)&Bs[p][bC + 2][kk] = pack2bf(bReg[it][0].z, bReg[it][1].z);
            *(unsigned*)&Bs[p][bC + 3][kk] = pack2bf(bReg[it][0].w, bReg[it][1].w);
        }
    };

    loadGlobal(0);
    storeLds(0);
    __syncthreads();

    int p = 0;
    for (int kb = 0; kb < K; kb += 64) {
        const bool more = (kb + 64 < K);
        if (more) loadGlobal(kb + 64);   // prefetch next stage into registers

        // 8 WMMAs on the resident stage
#pragma unroll
        for (int kk = 0; kk < 64; kk += 32) {
            bf16x16 af[2], bfr[2];
#pragma unroll
            for (int ms = 0; ms < 2; ++ms) {
                int mr = wm * 32 + ms * 16 + nl;
                af[ms] = load_frag_split(&As[p][mr][kk + klbA],
                                         &As[p][mr][kk + 16 + klbA]);
            }
#pragma unroll
            for (int ns = 0; ns < 2; ++ns) {
                int nr = wn * 32 + ns * 16 + nl;
                bfr[ns] = load_frag_split(&Bs[p][nr][kk + klbA],
                                          &Bs[p][nr][kk + 16 + klbA]);
            }
#pragma unroll
            for (int ms = 0; ms < 2; ++ms)
#pragma unroll
                for (int ns = 0; ns < 2; ++ns)
                    acc[ms][ns] = wmma_bf16(af[ms], bfr[ns], acc[ms][ns]);
        }

        if (more) storeLds(p ^ 1);       // fill the other buffer
        __syncthreads();
        p ^= 1;
    }

    const int rowAdd = (lane < 16) ? 0 : 8;
#pragma unroll
    for (int ms = 0; ms < 2; ++ms)
#pragma unroll
        for (int ns = 0; ns < 2; ++ns) {
            int col = col0 + wn * 32 + ns * 16 + nl;
            float bv = bias[col];
#pragma unroll
            for (int r = 0; r < 8; ++r) {
                int row = row0 + wm * 32 + ms * 16 + r + rowAdd;
                float v = acc[ms][ns][r] + bv;
                if (RELU) v = fmaxf(v, 0.0f);
                C[(size_t)row * N + col] = v;
            }
        }
}

// ---------------------------------------------------------------------------
// Flash attention: O = softmax(mask(Q K^T / sqrt(dk))) V  per (b,h).
// Q: [B*Sq, DM] (head h at cols h*64..), K/V: [B*Sk, DM]. 128 thr = 4 waves,
// each wave owns 16 query rows; 32-key blocks; online softmax; WMMA for both
// QK^T and P*V. masked!=0 -> causal + pad(tgt) mask (self-attention).
// ---------------------------------------------------------------------------
__global__ __launch_bounds__(128) void hl_flash_attn(
    const float* __restrict__ Q, const float* __restrict__ Kg,
    const float* __restrict__ Vg, float* __restrict__ O,
    const int* __restrict__ tgt, int Sq, int Sk, int masked)
{
    __shared__ unsigned short Ks[32][64];      // [key][d]
    __shared__ unsigned short Vs[64][32];      // transposed: [d][key]
    __shared__ unsigned short Ps[4][16][32];   // per-wave P staging [m][key]

    const int tid  = threadIdx.x;
    const int lane = tid & 31;
    const int wave = tid >> 5;
    const int bh   = blockIdx.y;
    const int b    = bh >> 4;
    const int h    = bh & 15;
    const int q0   = blockIdx.x * 64 + wave * 16;

    const int nl     = lane & 15;
    const int rowAdd = (lane < 16) ? 0 : 8;
    const int klbA   = (lane < 16) ? 0 : 8;
    const int klbB   = (lane < 16) ? 0 : 16;

    // Q fragments (A-matrix layout), pre-scaled by 1/sqrt(64)
    bf16x16 qf[2];
    {
        const float* qp = Q + ((size_t)(b * Sq + q0 + nl)) * DM + h * DK;
#pragma unroll
        for (int t = 0; t < 2; ++t) {
            BF16FragU f;
#pragma unroll
            for (int i = 0; i < 8; ++i) {
                f.u[i]     = f2b(qp[t * 32 + klbA + i] * 0.125f);
                f.u[i + 8] = f2b(qp[t * 32 + 16 + klbA + i] * 0.125f);
            }
            qf[t] = f.b;
        }
    }

    float mrun[8], lrun[8];
    f32x8 oacc[4];
    f32x8 zero = {};
#pragma unroll
    for (int r = 0; r < 8; ++r) { mrun[r] = -3.0e38f; lrun[r] = 0.0f; }
#pragma unroll
    for (int t = 0; t < 4; ++t) oacc[t] = zero;

    int rowpad[8];
#pragma unroll
    for (int r = 0; r < 8; ++r) {
        int qpos = q0 + r + rowAdd;
        rowpad[r] = masked ? (tgt[b * Sq + qpos] == 0) : 0;
    }

    const int ldK = (tid >> 4) * 2;  // 0,2,..,14 (+16 on 2nd iter)
    const int ldC = (tid & 15) * 4;  // 0..60

    for (int kb = 0; kb < Sk; kb += 32) {
        __syncthreads();
        // K tile 32x64 natural, V tile stored transposed [d][key];
        // loads are k-row pairs so V stores pack 2 bf16 per dword.
#pragma unroll
        for (int it = 0; it < 2; ++it) {
            int kk = ldK + it * 16;
            const size_t rbase = (size_t)(b * Sk + kb + kk) * DM + h * DK + ldC;
            const float4 k0 = *(const float4*)(Kg + rbase);
            const float4 k1 = *(const float4*)(Kg + rbase + DM);
            uint2 pk0; pk0.x = pack2bf(k0.x, k0.y); pk0.y = pack2bf(k0.z, k0.w);
            uint2 pk1; pk1.x = pack2bf(k1.x, k1.y); pk1.y = pack2bf(k1.z, k1.w);
            *(uint2*)&Ks[kk][ldC]     = pk0;
            *(uint2*)&Ks[kk + 1][ldC] = pk1;
            const float4 v0 = *(const float4*)(Vg + rbase);
            const float4 v1 = *(const float4*)(Vg + rbase + DM);
            *(unsigned*)&Vs[ldC + 0][kk] = pack2bf(v0.x, v1.x);
            *(unsigned*)&Vs[ldC + 1][kk] = pack2bf(v0.y, v1.y);
            *(unsigned*)&Vs[ldC + 2][kk] = pack2bf(v0.z, v1.z);
            *(unsigned*)&Vs[ldC + 3][kk] = pack2bf(v0.w, v1.w);
        }
        __syncthreads();

        // scores: two 16x16 tiles covering 32 keys
        f32x8 s[2];
#pragma unroll
        for (int ns = 0; ns < 2; ++ns) {
            int key = ns * 16 + nl;
            bf16x16 kf0 = load_frag_split(&Ks[key][klbA], &Ks[key][16 + klbA]);
            bf16x16 kf1 = load_frag_split(&Ks[key][32 + klbA], &Ks[key][48 + klbA]);
            f32x8 z = {};
            s[ns] = wmma_bf16(qf[0], kf0, z);
            s[ns] = wmma_bf16(qf[1], kf1, s[ns]);
        }
        if (masked) {
#pragma unroll
            for (int ns = 0; ns < 2; ++ns) {
                int kpos = kb + ns * 16 + nl;
#pragma unroll
                for (int r = 0; r < 8; ++r) {
                    int qpos = q0 + r + rowAdd;
                    if (rowpad[r] || kpos > qpos) s[ns][r] = -1.0e9f;
                }
            }
        }
        // online softmax (row reductions within 16-lane halves)
#pragma unroll
        for (int r = 0; r < 8; ++r) {
            float mx = fmaxf(s[0][r], s[1][r]);
            mx = fmaxf(mx, __shfl_xor(mx, 1, 16));
            mx = fmaxf(mx, __shfl_xor(mx, 2, 16));
            mx = fmaxf(mx, __shfl_xor(mx, 4, 16));
            mx = fmaxf(mx, __shfl_xor(mx, 8, 16));
            float mnew = fmaxf(mrun[r], mx);
            float p0 = __expf(s[0][r] - mnew);
            float p1 = __expf(s[1][r] - mnew);
            float rs = p0 + p1;
            rs += __shfl_xor(rs, 1, 16);
            rs += __shfl_xor(rs, 2, 16);
            rs += __shfl_xor(rs, 4, 16);
            rs += __shfl_xor(rs, 8, 16);
            float corr = __expf(mrun[r] - mnew);
            lrun[r] = lrun[r] * corr + rs;
            mrun[r] = mnew;
#pragma unroll
            for (int t = 0; t < 4; ++t) oacc[t][r] *= corr;
            Ps[wave][r + rowAdd][nl]      = f2b(p0);
            Ps[wave][r + rowAdd][16 + nl] = f2b(p1);
        }
        // O += P @ V (contraction over 32 keys, N = 64 dk in 4 tiles)
        bf16x16 pf = load_frag_split(&Ps[wave][nl][klbA], &Ps[wave][nl][16 + klbA]);
#pragma unroll
        for (int t = 0; t < 4; ++t) {
            bf16x16 vf = load_frag_split(&Vs[t * 16 + nl][klbA], &Vs[t * 16 + nl][16 + klbA]);
            oacc[t] = wmma_bf16(pf, vf, oacc[t]);
        }
    }

#pragma unroll
    for (int t = 0; t < 4; ++t)
#pragma unroll
        for (int r = 0; r < 8; ++r) {
            int qpos = q0 + r + rowAdd;
            O[((size_t)(b * Sq + qpos)) * DM + h * DK + t * 16 + nl] =
                oacc[t][r] / lrun[r];
        }
}

// ---------------------------------------------------------------------------
// x = LayerNorm(x + a) * g + b   (row = blockIdx.x, row length 1024)
// ---------------------------------------------------------------------------
__global__ __launch_bounds__(256) void hl_add_ln(
    float* __restrict__ x, const float* __restrict__ a,
    const float* __restrict__ g, const float* __restrict__ bta)
{
    __shared__ float red[256];
    const int row = blockIdx.x;
    const int tid = threadIdx.x;
    const size_t base = (size_t)row * DM;

    float v[4];
    float s = 0.0f;
#pragma unroll
    for (int k = 0; k < 4; ++k) {
        int i = tid + k * 256;
        v[k] = x[base + i] + a[base + i];
        s += v[k];
    }
    red[tid] = s; __syncthreads();
    for (int off = 128; off > 0; off >>= 1) {
        if (tid < off) red[tid] += red[tid + off];
        __syncthreads();
    }
    float mu = red[0] * (1.0f / DM);
    __syncthreads();
    float s2 = 0.0f;
#pragma unroll
    for (int k = 0; k < 4; ++k) { float d = v[k] - mu; s2 += d * d; }
    red[tid] = s2; __syncthreads();
    for (int off = 128; off > 0; off >>= 1) {
        if (tid < off) red[tid] += red[tid + off];
        __syncthreads();
    }
    float var = red[0] * (1.0f / DM);
    float rstd = rsqrtf(var + 1e-5f);
#pragma unroll
    for (int k = 0; k < 4; ++k) {
        int i = tid + k * 256;
        x[base + i] = (v[k] - mu) * rstd * g[i] + bta[i];
    }
}

// ---------------------------------------------------------------------------
// Positional encoding helpers + embedding
// ---------------------------------------------------------------------------
__device__ __forceinline__ float pe_val(int s, int d) {
    float i2 = (float)(d & ~1);
    float freq = __expf(i2 * (-9.210340371976184f / (float)DM)); // -ln(10000)
    float ang = (float)s * freq;
    return (d & 1) ? __cosf(ang) : __sinf(ang);
}

__global__ __launch_bounds__(256) void hl_embed_pe(
    const int* __restrict__ tgt, const float* __restrict__ emb,
    float* __restrict__ X)
{
    int idx = blockIdx.x * 256 + threadIdx.x;  // [B*TGTL*DM)
    int d = idx & (DM - 1);
    int bs = idx >> 10;          // b*TGTL + s
    int s = bs & (TGTL - 1);
    int tok = tgt[bs];
    X[idx] = emb[(size_t)tok * DM + d] + pe_val(s, d);
}

__global__ __launch_bounds__(256) void hl_src_pe(
    const float* __restrict__ src, float* __restrict__ ENC)
{
    int idx = blockIdx.x * 256 + threadIdx.x;  // [B*SRCL*DM)
    int d = idx & (DM - 1);
    int bs = idx >> 10;
    int s = bs & (SRCL - 1);
    ENC[idx] = src[idx] + pe_val(s, d);
}

// ---------------------------------------------------------------------------
// Host launcher
// ---------------------------------------------------------------------------
extern "C" void kernel_launch(void* const* d_in, const int* in_sizes, int n_in,
                              void* d_out, int out_size, void* d_ws, size_t ws_size,
                              hipStream_t stream)
{
    const float* src  = (const float*)d_in[0];
    const int*   tgt  = (const int*)  d_in[1];
    const float* emb  = (const float*)d_in[2];
    const float* fc_w = (const float*)d_in[3];
    const float* fc_b = (const float*)d_in[4];
    const float* sa_wq = (const float*)d_in[5];
    const float* sa_bq = (const float*)d_in[6];
    const float* sa_wk = (const float*)d_in[7];
    const float* sa_bk = (const float*)d_in[8];
    const float* sa_wv = (const float*)d_in[9];
    const float* sa_bv = (const float*)d_in[10];
    const float* sa_wo = (const float*)d_in[11];
    const float* sa_bo = (const float*)d_in[12];
    const float* ca_wq = (const float*)d_in[13];
    const float* ca_bq = (const float*)d_in[14];
    const float* ca_wk = (const float*)d_in[15];
    const float* ca_bk = (const float*)d_in[16];
    const float* ca_wv = (const float*)d_in[17];
    const float* ca_bv = (const float*)d_in[18];
    const float* ca_wo = (const float*)d_in[19];
    const float* ca_bo = (const float*)d_in[20];
    const float* w1  = (const float*)d_in[21];
    const float* b1  = (const float*)d_in[22];
    const float* w2  = (const float*)d_in[23];
    const float* b2  = (const float*)d_in[24];
    const float* n1g = (const float*)d_in[25];
    const float* n2g = (const float*)d_in[26];
    const float* n3g = (const float*)d_in[27];
    const float* n1b = (const float*)d_in[28];
    const float* n2b = (const float*)d_in[29];
    const float* n3b = (const float*)d_in[30];

    const int Mtgt = BB * TGTL;   // 2048
    const int Msrc = BB * SRCL;   // 1024

    // workspace layout (f32): 21M floats = 84 MB
    float* W = (float*)d_ws;
    size_t off = 0;
    float* X   = W + off; off += (size_t)Mtgt * DM;   // current activations
    float* ENC = W + off; off += (size_t)Msrc * DM;   // encoder + PE
    float* Qb  = W + off; off += (size_t)Mtgt * DM;
    float* Kb  = W + off; off += (size_t)Mtgt * DM;
    float* Vb  = W + off; off += (size_t)Mtgt * DM;
    float* Ab  = W + off; off += (size_t)Mtgt * DM;   // attention output
    float* Tb  = W + off; off += (size_t)Mtgt * DM;   // proj / ffn2 output
    float* Fb  = W + off; off += (size_t)Mtgt * FFD;  // ffn hidden

    dim3 blk(256);
    hl_embed_pe<<<(Mtgt * DM) / 256, blk, 0, stream>>>(tgt, emb, X);
    hl_src_pe  <<<(Msrc * DM) / 256, blk, 0, stream>>>(src, ENC);

    auto gemm = [&](const float* A, const float* Wm, const float* bias,
                    float* C, int M, int N, int K, bool relu) {
        dim3 grid(N / 64, M / 128);
        if (relu)
            hl_gemm_bias<1><<<grid, 256, 0, stream>>>(A, Wm, bias, C, M, N, K);
        else
            hl_gemm_bias<0><<<grid, 256, 0, stream>>>(A, Wm, bias, C, M, N, K);
    };

    for (int l = 0; l < NL; ++l) {
        const size_t wo = (size_t)l * DM * DM;
        const size_t bo = (size_t)l * DM;

        // ---- self-attention ----
        gemm(X, sa_wq + wo, sa_bq + bo, Qb, Mtgt, DM, DM, false);
        gemm(X, sa_wk + wo, sa_bk + bo, Kb, Mtgt, DM, DM, false);
        gemm(X, sa_wv + wo, sa_bv + bo, Vb, Mtgt, DM, DM, false);
        hl_flash_attn<<<dim3(TGTL / 64, BB * NH), 128, 0, stream>>>(
            Qb, Kb, Vb, Ab, tgt, TGTL, TGTL, /*masked=*/1);
        gemm(Ab, sa_wo + wo, sa_bo + bo, Tb, Mtgt, DM, DM, false);
        hl_add_ln<<<Mtgt, 256, 0, stream>>>(X, Tb, n1g + bo, n1b + bo);

        // ---- cross-attention ----
        gemm(X,   ca_wq + wo, ca_bq + bo, Qb, Mtgt, DM, DM, false);
        gemm(ENC, ca_wk + wo, ca_bk + bo, Kb, Msrc, DM, DM, false);
        gemm(ENC, ca_wv + wo, ca_bv + bo, Vb, Msrc, DM, DM, false);
        hl_flash_attn<<<dim3(TGTL / 64, BB * NH), 128, 0, stream>>>(
            Qb, Kb, Vb, Ab, nullptr, TGTL, SRCL, /*masked=*/0);
        gemm(Ab, ca_wo + wo, ca_bo + bo, Tb, Mtgt, DM, DM, false);
        hl_add_ln<<<Mtgt, 256, 0, stream>>>(X, Tb, n2g + bo, n2b + bo);

        // ---- FFN ----
        gemm(X,  w1 + (size_t)l * DM * FFD, b1 + (size_t)l * FFD, Fb, Mtgt, FFD, DM, true);
        gemm(Fb, w2 + (size_t)l * FFD * DM, b2 + bo,              Tb, Mtgt, DM, FFD, false);
        hl_add_ln<<<Mtgt, 256, 0, stream>>>(X, Tb, n3g + bo, n3b + bo);
    }

    // ---- final vocab projection -> d_out [B, TGTL, VOC] ----
    gemm(X, fc_w, fc_b, (float*)d_out, Mtgt, VOC, DM, false);
}
